// TPALSTM_RES_7980049236136
// MI455X (gfx1250) — matrix operand order repaired and checked by reading, not compiled, not run on hardware
//
#include <hip/hip_runtime.h>
#include <hip/hip_bf16.h>

// ---------------------------------------------------------------------------
// Types for CDNA5 WMMA (wave32, v_wmma_f32_16x16x32_bf16)
// ---------------------------------------------------------------------------
typedef __bf16 bf16;
typedef __attribute__((ext_vector_type(8)))  bf16  v8bf;
typedef __attribute__((ext_vector_type(16))) bf16  v16bf;
typedef __attribute__((ext_vector_type(8)))  float v8f;

static __device__ __forceinline__ v16bf cat8(v8bf lo, v8bf hi) {
  return __builtin_shufflevector(lo, hi, 0,1,2,3,4,5,6,7,8,9,10,11,12,13,14,15);
}

// A-matrix fragment (16x32 bf16), ISA dense-A layout:
// lane<16 holds row (lane&15), K in {k0..k0+7} U {k0+16..k0+23};
// lane>=16 same row, K pattern shifted by +8.
static __device__ __forceinline__ v16bf load_a_frag(const bf16* base, int ld,
                                                    int row0, int k0, int lane) {
  int row  = row0 + (lane & 15);
  int half = (lane >> 4) & 1;
  const bf16* p = base + row * ld + k0 + half * 8;
  v8bf lo = *reinterpret_cast<const v8bf*>(p);
  v8bf hi = *reinterpret_cast<const v8bf*>(p + 16);
  return cat8(lo, hi);
}

// B-matrix fragment (32x16 bf16) computing C[m,n] = sum_k A[m,k]*W[n,k]
// (W row-major [N,K]): lane holds col n=(lane&15); lanes<16 K={k0..k0+7} U {k0+16..k0+23},
// lanes>=16 same pattern shifted by +8 (matches A layout).
static __device__ __forceinline__ v16bf load_b_frag(const bf16* W, int ld,
                                                    int n0, int k0, int lane) {
  int n    = n0 + (lane & 15);
  int koff = k0 + ((lane >> 4) & 1) * 8;
  const bf16* p = W + n * ld + koff;
  v8bf lo = *reinterpret_cast<const v8bf*>(p);
  v8bf hi = *reinterpret_cast<const v8bf*>(p + 16);
  return cat8(lo, hi);
}

static __device__ __forceinline__ v8f wmma_bf16(v16bf a, v16bf b, v8f c) {
  return __builtin_amdgcn_wmma_f32_16x16x32_bf16(false, a, false, b,
                                                 (short)0, c, false, false);
}

static __device__ __forceinline__ float sigmoidf_(float x) {
  return 1.0f / (1.0f + __expf(-x));
}

// ---------------------------------------------------------------------------
// Model constants
// ---------------------------------------------------------------------------
#define BB   64
#define TT   168
#define FIN  32
#define HSZ  256
#define FN_  32
#define OUTD 24

// ---------------------------------------------------------------------------
// fp32 -> bf16 convert
// ---------------------------------------------------------------------------
__global__ __launch_bounds__(256) void f32_to_bf16_kernel(
    const float* __restrict__ src, bf16* __restrict__ dst, int n) {
  int i = blockIdx.x * 256 + threadIdx.x;
  if (i < n) dst[i] = (bf16)src[i];
}

// ---------------------------------------------------------------------------
// Input projection: xc = relu(x @ W_hid^T + b_hid), M=10752 N=256 K=32, bf16 out
// 8 waves/block, 1 WMMA tile per wave.
// ---------------------------------------------------------------------------
__global__ __launch_bounds__(256) void proj_relu_kernel(
    const bf16* __restrict__ X, const bf16* __restrict__ W,
    const float* __restrict__ bias, bf16* __restrict__ XC) {
  int wave = threadIdx.x >> 5, lane = threadIdx.x & 31;
  int tile = blockIdx.x * 8 + wave;
  int row0 = (tile >> 4) * 16;
  int n0   = (tile & 15) * 16;
  v8f acc = {0.f,0.f,0.f,0.f,0.f,0.f,0.f,0.f};
  v16bf a = load_a_frag(X, FIN, row0, 0, lane);
  v16bf b = load_b_frag(W, FIN, n0, 0, lane);
  acc = wmma_bf16(a, b, acc);
  int col  = n0 + (lane & 15);
  int half = (lane >> 4) & 1;
  float bv = bias[col];
#pragma unroll
  for (int r = 0; r < 8; ++r) {
    int row = row0 + r + half * 8;
    float v = acc[r] + bv;
    XC[(long)row * HSZ + col] = (bf16)(v > 0.f ? v : 0.f);
  }
}

// ---------------------------------------------------------------------------
// Big GEMM: C[r,n] = sum_k A[r,k] * W[n,k]   (A bf16 [M,K], W bf16 [N,K], C f32)
// Used for the hoisted input-path gate preactivations (M=10752,N=1024,K=256).
// ---------------------------------------------------------------------------
__global__ __launch_bounds__(256) void gemm_ih_kernel(
    const bf16* __restrict__ A, const bf16* __restrict__ W,
    float* __restrict__ C, int N, int K) {
  int wave = threadIdx.x >> 5, lane = threadIdx.x & 31;
  int tile = blockIdx.x * 8 + wave;
  int ntn  = N >> 4;
  int row0 = (tile / ntn) * 16;
  int n0   = (tile % ntn) * 16;
  v8f acc = {0.f,0.f,0.f,0.f,0.f,0.f,0.f,0.f};
  for (int k0 = 0; k0 < K; k0 += 32) {
    v16bf a = load_a_frag(A, K, row0, k0, lane);
    v16bf b = load_b_frag(W, K, n0, k0, lane);
    acc = wmma_bf16(a, b, acc);
  }
  int col  = n0 + (lane & 15);
  int half = (lane >> 4) & 1;
#pragma unroll
  for (int r = 0; r < 8; ++r) {
    int row = row0 + r + half * 8;
    C[(long)row * N + col] = acc[r];
  }
}

// ---------------------------------------------------------------------------
// Persistent single-WGP LSTM recurrence. 1024 threads = 32 waves.
// h kept in LDS (bf16, padded stride 264 -> conflict-free b128 A-frag reads),
// c kept in registers. Each wave owns 2 (row-block,col-block) combos; per combo
// it computes the matching i/f/g/o 16x16 tiles (4 WMMA x 8 K-chunks).
// G = precomputed x-path preactivations. g_tmajor selects G row indexing.
// ---------------------------------------------------------------------------
__global__ __launch_bounds__(1024) void lstm_recur_kernel(
    const float* __restrict__ G, int g_tmajor,
    const bf16* __restrict__ Whh,
    const float* __restrict__ b_ih, const float* __restrict__ b_hh,
    bf16* __restrict__ hs_bf16, float* __restrict__ hs_f32) {
  const int HB_LD = 264;
  __shared__ bf16 hbuf[BB * HB_LD];

  int tid  = threadIdx.x;
  int wave = tid >> 5, lane = tid & 31;
  int half = (lane >> 4) & 1;
  int lan  = lane & 15;

  for (int i = tid; i < BB * HB_LD; i += 1024) hbuf[i] = (bf16)0.0f;
  __syncthreads();

  float cst[2][8];
  float hnew[2][8];
#pragma unroll
  for (int ci = 0; ci < 2; ++ci)
#pragma unroll
    for (int r = 0; r < 8; ++r) cst[ci][r] = 0.f;

  for (int t = 0; t < TT; ++t) {
#pragma unroll
    for (int ci = 0; ci < 2; ++ci) {
      int k  = wave * 2 + ci;
      int rb = k >> 4, cb = k & 15;
      int row0 = rb * 16;
      int col  = cb * 16 + lan;
      v8f acc[4];
      // init accumulators from hoisted x-path preactivations + bias
#pragma unroll
      for (int g = 0; g < 4; ++g) {
        int gc = g * HSZ + col;
        float bias = b_ih[gc] + b_hh[gc];
#pragma unroll
        for (int r = 0; r < 8; ++r) {
          int row = row0 + r + half * 8;
          long gi = g_tmajor ? ((long)(t * BB + row) * 1024 + gc)
                             : ((long)(row * TT + t) * 1024 + gc);
          acc[g][r] = G[gi] + bias;
        }
      }
      // h @ Whh^T via WMMA, A from LDS, B from global (WGP$/L2 resident)
#pragma unroll
      for (int k0 = 0; k0 < HSZ; k0 += 32) {
        v16bf a = load_a_frag(hbuf, HB_LD, row0, k0, lane);
#pragma unroll
        for (int g = 0; g < 4; ++g) {
          v16bf b = load_b_frag(Whh, HSZ, g * HSZ + cb * 16, k0, lane);
          acc[g] = wmma_bf16(a, b, acc[g]);
        }
      }
      // activations + cell update (this lane owns c for these elements)
#pragma unroll
      for (int r = 0; r < 8; ++r) {
        float ig = sigmoidf_(acc[0][r]);
        float fg = sigmoidf_(acc[1][r]);
        float gg = tanhf(acc[2][r]);
        float og = sigmoidf_(acc[3][r]);
        float c  = fg * cst[ci][r] + ig * gg;
        cst[ci][r]  = c;
        hnew[ci][r] = og * tanhf(c);
      }
    }
    __syncthreads();  // all waves done reading old h from LDS
#pragma unroll
    for (int ci = 0; ci < 2; ++ci) {
      int k  = wave * 2 + ci;
      int rb = k >> 4, cb = k & 15;
      int col = cb * 16 + lan;
#pragma unroll
      for (int r = 0; r < 8; ++r) {
        int row  = rb * 16 + r + half * 8;
        float hv = hnew[ci][r];
        bf16 hb  = (bf16)hv;
        hbuf[row * HB_LD + col] = hb;
        hs_bf16[((long)t * BB + row) * HSZ + col] = hb;
        if (hs_f32) hs_f32[((long)t * BB + row) * HSZ + col] = hv;
      }
    }
    __syncthreads();  // new h visible before next step
  }
}

// ---------------------------------------------------------------------------
// BN statistics pass: per-channel sum / sumsq of conv1(H) without storing c1.
// hs1 layout [T,B,HS]; H(b,t,j) = relu(hs1[t,b,j]), t in [0,167).
// ---------------------------------------------------------------------------
__global__ __launch_bounds__(256) void zero_stats_kernel(float* stats) {
  if (threadIdx.x < 64) stats[threadIdx.x] = 0.f;
}

__global__ __launch_bounds__(256) void bn_stats_kernel(
    const float* __restrict__ hs1, const float* __restrict__ w1,
    const float* __restrict__ b1, float* __restrict__ stats) {
  __shared__ float sw[288], sb[32], ssum[32], ssq[32];
  int tid = threadIdx.x;
  for (int i = tid; i < 288; i += 256) sw[i] = w1[i];
  if (tid < 32) { sb[tid] = b1[tid]; ssum[tid] = 0.f; ssq[tid] = 0.f; }
  __syncthreads();

  int idx = blockIdx.x * 256 + tid;          // 64*167*256 positions exactly
  int b   = idx / (167 * 256);
  int rem = idx - b * (167 * 256);
  int t   = rem >> 8;
  int j   = rem & 255;

  float Hn[9];
#pragma unroll
  for (int dt = 0; dt < 3; ++dt)
#pragma unroll
    for (int dj = 0; dj < 3; ++dj) {
      int ti = t + dt - 1, ji = j + dj - 1;
      float v = 0.f;
      if (ti >= 0 && ti < 167 && ji >= 0 && ji < 256) {
        v = hs1[((long)ti * BB + b) * HSZ + ji];
        v = v > 0.f ? v : 0.f;
      }
      Hn[dt * 3 + dj] = v;
    }
  for (int f = 0; f < 32; ++f) {
    float a = sb[f];
#pragma unroll
    for (int q = 0; q < 9; ++q) a += sw[f * 9 + q] * Hn[q];
    atomicAdd(&ssum[f], a);
    atomicAdd(&ssq[f], a * a);
  }
  __syncthreads();
  if (tid < 32) {
    atomicAdd(&stats[tid], ssum[tid]);
    atomicAdd(&stats[32 + tid], ssq[tid]);
  }
}

__global__ __launch_bounds__(64) void bn_finalize_kernel(
    float* __restrict__ stats, const float* __restrict__ bn_g,
    const float* __restrict__ bn_b) {
  int f = threadIdx.x;
  if (f < 32) {
    const float N = 64.0f * 167.0f * 256.0f;
    float mean = stats[f] / N;
    float var  = stats[32 + f] / N - mean * mean;
    float inv  = rsqrtf(var + 1e-5f);
    float sc   = bn_g[f] * inv;
    stats[64 + f] = sc;                    // scale
    stats[96 + f] = bn_b[f] - mean * sc;   // shift
  }
}

// ---------------------------------------------------------------------------
// Fused conv1 -> BN -> relu -> conv2 (+ residual H), tiled 8x32 per block,
// c1 tile staged in LDS so the 350MB c1 tensor is never materialized.
// ---------------------------------------------------------------------------
__global__ __launch_bounds__(256) void conv2_fused_kernel(
    const float* __restrict__ hs1, const float* __restrict__ w1,
    const float* __restrict__ b1, const float* __restrict__ stats,
    const float* __restrict__ w2, const float* __restrict__ b2,
    float* __restrict__ c2) {
  __shared__ float sH[12][36];
  __shared__ float sc1[32][10][34];
  __shared__ float sw1[288], sw2[288], sb1[32], ssc[32], ssh[32];

  int b  = blockIdx.x;
  int t0 = blockIdx.y * 8;
  int j0 = blockIdx.z * 32;
  int tid = threadIdx.x;

  for (int i = tid; i < 288; i += 256) { sw1[i] = w1[i]; sw2[i] = w2[i]; }
  if (tid < 32) { sb1[tid] = b1[tid]; ssc[tid] = stats[64 + tid]; ssh[tid] = stats[96 + tid]; }
  for (int i = tid; i < 12 * 36; i += 256) {
    int th = i / 36, jh = i % 36;
    int t = t0 - 2 + th, j = j0 - 2 + jh;
    float v = 0.f;
    if (t >= 0 && t < 167 && j >= 0 && j < 256) {
      v = hs1[((long)t * BB + b) * HSZ + j];
      v = v > 0.f ? v : 0.f;
    }
    sH[th][jh] = v;
  }
  __syncthreads();

  for (int i = tid; i < 32 * 10 * 34; i += 256) {
    int f = i / 340, r = i % 340;
    int th = r / 34, jh = r % 34;
    int t = t0 - 1 + th, j = j0 - 1 + jh;
    float v = 0.f;
    if (t >= 0 && t < 167 && j >= 0 && j < 256) {
      float a = sb1[f];
#pragma unroll
      for (int dt = 0; dt < 3; ++dt)
#pragma unroll
        for (int dj = 0; dj < 3; ++dj)
          a += sw1[f * 9 + dt * 3 + dj] * sH[th + dt][jh + dj];
      a = ssc[f] * a + ssh[f];
      v = a > 0.f ? a : 0.f;
    }
    sc1[f][th][jh] = v;
  }
  __syncthreads();

  int th = tid >> 5, jh = tid & 31;
  int t = t0 + th, j = j0 + jh;
  if (t < 167) {
    float a = b2[0];
    for (int f = 0; f < 32; ++f)
#pragma unroll
      for (int dt = 0; dt < 3; ++dt)
#pragma unroll
        for (int dj = 0; dj < 3; ++dj)
          a += sw2[f * 9 + dt * 3 + dj] * sc1[f][th + dt][jh + dj];
    a += sH[th + 2][jh + 2];  // residual H
    c2[((long)b * 167 + t) * 256 + j] = a;
  }
}

// ---------------------------------------------------------------------------
// conv3 einsum: cv[b, f*256+j] = relu(b3[f] + sum_t w3[f,t] * c2[b,t,j])
// ---------------------------------------------------------------------------
__global__ __launch_bounds__(256) void conv3_kernel(
    const float* __restrict__ c2, const float* __restrict__ w3,
    const float* __restrict__ b3, float* __restrict__ cv) {
  int idx = blockIdx.x * 256 + threadIdx.x;  // 64*32*256 exactly
  int b = idx >> 13;
  int rem = idx & 8191;
  int f = rem >> 8;
  int j = rem & 255;
  float a = b3[f];
  const float* cp = c2 + (long)b * 167 * 256 + j;
  const float* wp = w3 + f * 167;
  for (int t = 0; t < 167; ++t) a += wp[t] * cp[t * 256];
  cv[(long)b * 8192 + rem] = a > 0.f ? a : 0.f;
}

// ---------------------------------------------------------------------------
// Attention head + final linears. One block per batch row.
// cvr(h2,f2) == cv flat [h2*32+f2] (reshape, not transpose).
// ---------------------------------------------------------------------------
__global__ __launch_bounds__(256) void head_kernel(
    const float* __restrict__ hs1, const float* __restrict__ cv,
    const float* __restrict__ lin1_w, const float* __restrict__ lin1_b,
    const float* __restrict__ lin2_w, const float* __restrict__ lin2_b,
    const float* __restrict__ out_w, const float* __restrict__ out_b,
    float* __restrict__ out) {
  __shared__ float htts[256], wvec[32], vacc[32], nh[256];
  int b = blockIdx.x, j = threadIdx.x;
  htts[j] = hs1[((long)(TT - 1) * BB + b) * HSZ + j];
  __syncthreads();
  if (j < 32) {
    float a = lin1_b[j];
    for (int k = 0; k < 256; ++k) a += htts[k] * lin1_w[j * 256 + k];
    wvec[j] = a;
    vacc[j] = 0.f;
  }
  __syncthreads();
  const float* cvb = cv + (long)b * 8192;
  float s = 0.f;
#pragma unroll
  for (int f = 0; f < 32; ++f) s += cvb[j * 32 + f] * wvec[f];
  float alpha = 1.f / (1.f + __expf(-s));
  for (int f = 0; f < 32; ++f) atomicAdd(&vacc[f], alpha * cvb[j * 32 + f]);
  __syncthreads();
  float a = lin2_b[j];
  for (int k = 0; k < 256; ++k) a += htts[k] * lin2_w[j * 288 + k];
  for (int k = 0; k < 32; ++k) a += vacc[k] * lin2_w[j * 288 + 256 + k];
  nh[j] = a;
  __syncthreads();
  if (j < OUTD) {
    float o = out_b[j];
    for (int n = 0; n < 256; ++n) o += nh[n] * out_w[j * 256 + n];
    out[b * OUTD + j] = o;
  }
}

// ---------------------------------------------------------------------------
// Launch: graph-capture safe, all on `stream`, workspace-only scratch.
// ---------------------------------------------------------------------------
extern "C" void kernel_launch(void* const* d_in, const int* in_sizes, int n_in,
                              void* d_out, int out_size, void* d_ws, size_t ws_size,
                              hipStream_t stream) {
  (void)in_sizes; (void)n_in; (void)out_size; (void)ws_size;
  const float* x       = (const float*)d_in[0];
  const float* W_hid   = (const float*)d_in[1];
  const float* b_hid   = (const float*)d_in[2];
  const float* W_ih    = (const float*)d_in[3];
  const float* W_hh    = (const float*)d_in[4];
  const float* b_ih    = (const float*)d_in[5];
  const float* b_hh    = (const float*)d_in[6];
  const float* conv1_w = (const float*)d_in[7];
  const float* conv1_b = (const float*)d_in[8];
  const float* bn_g    = (const float*)d_in[9];
  const float* bn_b    = (const float*)d_in[10];
  const float* conv2_w = (const float*)d_in[11];
  const float* conv2_b = (const float*)d_in[12];
  const float* conv3_w = (const float*)d_in[13];
  const float* conv3_b = (const float*)d_in[14];
  const float* lin1_w  = (const float*)d_in[15];
  const float* lin1_b  = (const float*)d_in[16];
  const float* lin2_w  = (const float*)d_in[17];
  const float* lin2_b  = (const float*)d_in[18];
  const float* out_w   = (const float*)d_in[19];
  const float* out_b   = (const float*)d_in[20];

  char* ws = (char*)d_ws;
  bf16*  x_bf    = (bf16*)(ws + 0);          // 344064 bf16
  bf16*  whid_bf = (bf16*)(ws + 688128);     // 8192 bf16
  bf16*  wih_bf  = (bf16*)(ws + 704512);     // 524288 bf16 (both layers)
  bf16*  whh_bf  = (bf16*)(ws + 1753088);    // 524288 bf16 (both layers)
  bf16*  xc_bf   = (bf16*)(ws + 2801664);    // 10752*256 bf16  [B,T,HS]
  bf16*  hs_bf   = (bf16*)(ws + 8306688);    // 10752*256 bf16  [T,B,HS]
  float* hs1     = (float*)(ws + 13811712);  // 10752*256 f32   [T,B,HS]
  float* G       = (float*)(ws + 24821760);  // 10752*1024 f32  (reused below)
  float* c2      = (float*)(ws + 24821760);  // 64*167*256 f32  (overlays G)
  float* cvb     = (float*)(ws + 35766272);  // 64*8192 f32     (overlays G)
  float* stats   = (float*)(ws + 68861952);  // 128 f32

  // bf16 conversions
  f32_to_bf16_kernel<<<1344, 256, 0, stream>>>(x, x_bf, 344064);
  f32_to_bf16_kernel<<<32,   256, 0, stream>>>(W_hid, whid_bf, 8192);
  f32_to_bf16_kernel<<<2048, 256, 0, stream>>>(W_ih, wih_bf, 524288);
  f32_to_bf16_kernel<<<2048, 256, 0, stream>>>(W_hh, whh_bf, 524288);

  // xc = relu(x @ W_hid^T + b)        (WMMA, full chip)
  proj_relu_kernel<<<1344, 256, 0, stream>>>(x_bf, whid_bf, b_hid, xc_bf);

  // layer 0: hoisted x-path GEMM, then persistent recurrence
  gemm_ih_kernel<<<5376, 256, 0, stream>>>(xc_bf, wih_bf, G, 1024, 256);
  lstm_recur_kernel<<<1, 1024, 0, stream>>>(G, 0, whh_bf, b_ih, b_hh,
                                            hs_bf, nullptr);
  // layer 1
  gemm_ih_kernel<<<5376, 256, 0, stream>>>(hs_bf, wih_bf + 262144, G, 1024, 256);
  lstm_recur_kernel<<<1, 1024, 0, stream>>>(G, 1, whh_bf + 262144,
                                            b_ih + 1024, b_hh + 1024,
                                            hs_bf, hs1);

  // conv head
  zero_stats_kernel<<<1, 64, 0, stream>>>(stats);
  bn_stats_kernel<<<10688, 256, 0, stream>>>(hs1, conv1_w, conv1_b, stats);
  bn_finalize_kernel<<<1, 64, 0, stream>>>(stats, bn_g, bn_b);
  conv2_fused_kernel<<<dim3(64, 21, 8), 256, 0, stream>>>(
      hs1, conv1_w, conv1_b, stats, conv2_w, conv2_b, c2);
  conv3_kernel<<<2048, 256, 0, stream>>>(c2, conv3_w, conv3_b, cvb);

  // attention + linear tail -> d_out (f32, 64x24)
  head_kernel<<<64, 256, 0, stream>>>(hs1, cvb, lin1_w, lin1_b,
                                      lin2_w, lin2_b, out_w, out_b,
                                      (float*)d_out);
}
